// MHA_aLiBi_74423193305530
// MI455X (gfx1250) — compile-verified, matrix operand order checked by reading, not run on hardware
//
#include <hip/hip_runtime.h>

#define N_HEADS 16
#define DMODEL  1024
#define HEADD   64
#define SEQ     2048
#define BATCH   2
#define MGEMM   (BATCH * SEQ)   // 4096
#define KGEMM   DMODEL          // 1024
#define NGEMM   DMODEL          // 1024

typedef __attribute__((ext_vector_type(16))) __bf16       v16bf;
typedef __attribute__((ext_vector_type(8)))  float        v8f;
typedef __attribute__((ext_vector_type(8)))  __bf16       bf16x8;
typedef __attribute__((ext_vector_type(4)))  unsigned int u32x4;
typedef __attribute__((ext_vector_type(4)))  int          i32x4;

#if defined(__HIP_DEVICE_COMPILE__) && __has_builtin(__builtin_amdgcn_global_load_async_to_lds_b128)
#define HAVE_ASYNC_LDS 1
#else
#define HAVE_ASYNC_LDS 0
#endif

// Copy one 16-byte chunk global -> LDS. On CDNA5 this is an async DMA
// (GLOBAL_LOAD_ASYNC_TO_LDS_B128, tracked with ASYNCcnt); fallback is a
// synchronous VGPR round-trip.
__device__ __forceinline__ void async_copy16(const __bf16* __restrict__ g,
                                             __bf16* __restrict__ l) {
#if HAVE_ASYNC_LDS
  __builtin_amdgcn_global_load_async_to_lds_b128(
      (__attribute__((address_space(1))) i32x4*)g,
      (__attribute__((address_space(3))) i32x4*)l, 0, 0);
#else
  *reinterpret_cast<u32x4*>(l) = *reinterpret_cast<const u32x4*>(g);
#endif
}

__device__ __forceinline__ void wait_async() {
#if HAVE_ASYNC_LDS
  asm volatile("s_wait_asynccnt 0" ::: "memory");
#endif
}

// Build a 16-half A/B fragment per the CDNA5 16-bit layout:
// caller passes p = row_base + kbase (kbase = 0 for lanes 0-15, 8 for 16-31);
// halves 0..7 come from p[0..7], halves 8..15 from p[16..23].
__device__ __forceinline__ v16bf load_frag16(const __bf16* p) {
  bf16x8 lo = *reinterpret_cast<const bf16x8*>(p);
  bf16x8 hi = *reinterpret_cast<const bf16x8*>(p + 16);
  v16bf f;
#pragma unroll
  for (int e = 0; e < 8; ++e) { f[e] = lo[e]; f[e + 8] = hi[e]; }
  return f;
}

__global__ void cvt_f32_bf16(const float* __restrict__ in, __bf16* __restrict__ out, int n) {
  int i = blockIdx.x * blockDim.x + threadIdx.x;
  if (i < n) out[i] = (__bf16)in[i];
}

// C[m,n] = sum_k A[m,k] * W[n,k] + bias[n], fixed 4096x1024x1024.
// MODE 0: write bf16 head-split [B,H,S,hd]; MODE 1: write f32 row-major [M,N].
// LDS double-buffered; next K-slab staged via async-to-LDS DMA while the
// current slab feeds the WMMAs.
template <int MODE>
__global__ __launch_bounds__(256)
void gemm_bf16(const __bf16* __restrict__ A, const __bf16* __restrict__ W,
               const float* __restrict__ bias, __bf16* __restrict__ outb,
               float* __restrict__ outf)
{
  __shared__ __bf16 At[2][128 * 40];   // 128 rows x 32 halves, stride 40 (80B, 16B-aligned)
  __shared__ __bf16 Bt[2][64 * 40];

  const int tid  = threadIdx.x;
  const int lane = tid & 31;
  const int wave = tid >> 5;
  const int wm   = wave & 3;           // 4 waves along M (4*32 = 128)
  const int wn   = wave >> 2;          // 2 waves along N (2*32 = 64)
  const int m0   = blockIdx.x * 128;
  const int n0   = blockIdx.y * 64;
  const int lr   = lane & 15;
  const int kb   = (lane < 16) ? 0 : 8;
  const int mh   = (lane >= 16) ? 8 : 0;

  // Fixed per-thread staging chunks (16B each): A tile = 512 chunks (2/thread),
  // B tile = 256 chunks (1/thread).
  const int cr = tid >> 2;             // 0..63
  const int cc = (tid & 3) * 8;        // 0,8,16,24
  const __bf16* gA0 = A + (size_t)(m0 + cr) * KGEMM + cc;
  const __bf16* gA1 = A + (size_t)(m0 + 64 + cr) * KGEMM + cc;
  const __bf16* gB0 = W + (size_t)(n0 + cr) * KGEMM + cc;
  const int lo0 = cr * 40 + cc;
  const int lo1 = (64 + cr) * 40 + cc;

  v8f acc[2][2];
#pragma unroll
  for (int i = 0; i < 2; ++i)
#pragma unroll
    for (int j = 0; j < 2; ++j)
#pragma unroll
      for (int e = 0; e < 8; ++e) acc[i][j][e] = 0.0f;

  // prologue: stage slab 0 into buffer 0
  async_copy16(gA0, &At[0][lo0]);
  async_copy16(gA1, &At[0][lo1]);
  async_copy16(gB0, &Bt[0][lo0]);

#pragma unroll 1
  for (int kt = 0; kt < KGEMM / 32; ++kt) {
    const int p = kt & 1;
    wait_async();
    __syncthreads();
    if (kt + 1 < KGEMM / 32) {          // stage next slab into the other buffer
      const int np = p ^ 1;
      const int ko = (kt + 1) * 32;
      async_copy16(gA0 + ko, &At[np][lo0]);
      async_copy16(gA1 + ko, &At[np][lo1]);
      async_copy16(gB0 + ko, &Bt[np][lo0]);
    }

    v16bf af[2], bf[2];
#pragma unroll
    for (int i = 0; i < 2; ++i)
      af[i] = load_frag16(&At[p][(wm * 32 + i * 16 + lr) * 40 + kb]);
#pragma unroll
    for (int j = 0; j < 2; ++j)
      bf[j] = load_frag16(&Bt[p][(wn * 32 + j * 16 + lr) * 40 + kb]);

#pragma unroll
    for (int i = 0; i < 2; ++i)
#pragma unroll
      for (int j = 0; j < 2; ++j)
        acc[i][j] = __builtin_amdgcn_wmma_f32_16x16x32_bf16(
            false, af[i], false, bf[j], (short)0, acc[i][j], false, false);
  }

#pragma unroll
  for (int i = 0; i < 2; ++i)
#pragma unroll
    for (int j = 0; j < 2; ++j)
#pragma unroll
      for (int e = 0; e < 8; ++e) {
        int m = m0 + wm * 32 + i * 16 + e + mh;
        int n = n0 + wn * 32 + j * 16 + lr;
        float v = acc[i][j][e] + bias[n];
        if (MODE == 0) {
          int b = m / SEQ, s = m % SEQ, h = n / HEADD, d = n % HEADD;
          outb[((size_t)(b * N_HEADS + h) * SEQ + s) * HEADD + d] = (__bf16)v;
        } else {
          outf[(size_t)m * NGEMM + n] = v;
        }
      }
}

// Flash-attention with ALiBi + causal mask. One block = (b,h) x 128 query rows.
__global__ __launch_bounds__(256)
void attn_alibi(const __bf16* __restrict__ Q, const __bf16* __restrict__ Kp,
                const __bf16* __restrict__ Vp, __bf16* __restrict__ O)
{
  __shared__ __bf16 Kt[64 * 72];        // K tile [j][d], stride 72 halves (144B)
  __shared__ __bf16 Vt[64 * 72];        // V tile transposed [d][j]
  __shared__ __bf16 Pb[8 * 16 * 64];    // per-wave P staging (16x64 each)

  const int tid  = threadIdx.x;
  const int lane = tid & 31;
  const int wave = tid >> 5;
  const int bh   = blockIdx.y;          // b*H + h
  const int h    = bh & (N_HEADS - 1);
  const int b    = bh >> 4;
  const int q0   = blockIdx.x * 128;
  const int lr   = lane & 15;
  const int kb   = (lane < 16) ? 0 : 8;
  const int mh   = (lane >= 16) ? 8 : 0;
  const float slope = exp2f(-(float)(h + 1));   // H=16 power of two -> 0.5^(h+1)

  const size_t headoff = (size_t)bh * SEQ * HEADD;
  const int qrow = q0 + wave * 16 + lr;

  // Fixed per-thread K-tile staging chunks: 64 rows x 8 chunks = 512 (2/thread)
  const int kr = tid >> 3;              // 0..31
  const int kc = (tid & 7) * 8;
  const __bf16* gK0 = Kp + headoff + (size_t)kr * HEADD + kc;
  const __bf16* gK1 = Kp + headoff + (size_t)(32 + kr) * HEADD + kc;
  const int klo0 = kr * 72 + kc;
  const int klo1 = (32 + kr) * 72 + kc;

  // Q fragments held in registers for the whole pass (hd=64 -> two K chunks)
  v16bf qf[2];
#pragma unroll
  for (int kd = 0; kd < 2; ++kd)
    qf[kd] = load_frag16(Q + headoff + (size_t)qrow * HEADD + kd * 32 + kb);

  v8f of[4];
  float mrow[8], lrow[8];
#pragma unroll
  for (int dn = 0; dn < 4; ++dn)
#pragma unroll
    for (int e = 0; e < 8; ++e) of[dn][e] = 0.0f;
#pragma unroll
  for (int e = 0; e < 8; ++e) { mrow[e] = -1e30f; lrow[e] = 0.0f; }

#pragma unroll 1
  for (int jb = 0; jb < q0 + 128; jb += 64) {   // causal: only key blocks <= diag
    __syncthreads();                            // previous iteration done reading tiles
    // K tile via async DMA to LDS
    async_copy16(gK0 + (size_t)jb * HEADD, &Kt[klo0]);
    async_copy16(gK1 + (size_t)jb * HEADD, &Kt[klo1]);
    // V tile, transposed into LDS so P@V B-fragments read contiguous j-runs
    for (int t = tid; t < 4096; t += 256) {
      int r = t >> 6, c = t & 63;
      Vt[c * 72 + r] = Vp[headoff + (size_t)(jb + r) * HEADD + c];
    }
    wait_async();
    __syncthreads();

    // S = Q K^T : 16x64 per wave
    v8f sf[4];
#pragma unroll
    for (int jn = 0; jn < 4; ++jn) {
#pragma unroll
      for (int e = 0; e < 8; ++e) sf[jn][e] = 0.0f;
#pragma unroll
      for (int kd = 0; kd < 2; ++kd) {
        v16bf bfrag = load_frag16(&Kt[(jn * 16 + lr) * 72 + kd * 32 + kb]);
        sf[jn] = __builtin_amdgcn_wmma_f32_16x16x32_bf16(
            false, qf[kd], false, bfrag, (short)0, sf[jn], false, false);
      }
    }

    // ALiBi + causal mask, per accumulator element
    const int iq = q0 + wave * 16 + mh;
#pragma unroll
    for (int jn = 0; jn < 4; ++jn)
#pragma unroll
      for (int e = 0; e < 8; ++e) {
        int i = iq + e;
        int j = jb + jn * 16 + lr;
        float v = sf[jn][e];
        sf[jn][e] = (j <= i) ? (v - slope * (float)(i - j)) : -1e9f;
      }

    // online softmax: row max across 16 lanes of each half-wave
    float nm[8];
#pragma unroll
    for (int e = 0; e < 8; ++e) {
      float v = sf[0][e];
#pragma unroll
      for (int jn = 1; jn < 4; ++jn) v = fmaxf(v, sf[jn][e]);
#pragma unroll
      for (int msk = 8; msk >= 1; msk >>= 1) v = fmaxf(v, __shfl_xor(v, msk, 32));
      nm[e] = v;
    }
    float scl[8];
#pragma unroll
    for (int e = 0; e < 8; ++e) {
      float mnew = fmaxf(mrow[e], nm[e]);
      scl[e] = __expf(mrow[e] - mnew);
      mrow[e] = mnew;
    }
    float rs[8];
#pragma unroll
    for (int e = 0; e < 8; ++e) rs[e] = 0.0f;
#pragma unroll
    for (int jn = 0; jn < 4; ++jn)
#pragma unroll
      for (int e = 0; e < 8; ++e) {
        float p = __expf(sf[jn][e] - mrow[e]);
        sf[jn][e] = p;
        rs[e] += p;
      }
#pragma unroll
    for (int e = 0; e < 8; ++e) {
#pragma unroll
      for (int msk = 8; msk >= 1; msk >>= 1) rs[e] += __shfl_xor(rs[e], msk, 32);
      lrow[e] = lrow[e] * scl[e] + rs[e];
    }
#pragma unroll
    for (int dn = 0; dn < 4; ++dn)
#pragma unroll
      for (int e = 0; e < 8; ++e) of[dn][e] *= scl[e];

    // stage P (C-layout) into per-wave LDS, re-read in A-fragment layout
    __bf16* pw = &Pb[wave * 16 * 64];
#pragma unroll
    for (int jn = 0; jn < 4; ++jn)
#pragma unroll
      for (int e = 0; e < 8; ++e)
        pw[(e + mh) * 64 + jn * 16 + lr] = (__bf16)sf[jn][e];
    asm volatile("s_wait_dscnt 0" ::: "memory");   // wave-local LDS RAW fence

    v16bf pa[2];
#pragma unroll
    for (int kj = 0; kj < 2; ++kj)
      pa[kj] = load_frag16(pw + lr * 64 + kj * 32 + kb);

    // O += P V  (B-fragments from transposed V tile)
#pragma unroll
    for (int dn = 0; dn < 4; ++dn)
#pragma unroll
      for (int kj = 0; kj < 2; ++kj) {
        v16bf bfrag = load_frag16(&Vt[(dn * 16 + lr) * 72 + kj * 32 + kb]);
        of[dn] = __builtin_amdgcn_wmma_f32_16x16x32_bf16(
            false, pa[kj], false, bfrag, (short)0, of[dn], false, false);
      }
  }

  // normalize and write merged-head bf16 output [B,S,D]
#pragma unroll
  for (int dn = 0; dn < 4; ++dn)
#pragma unroll
    for (int e = 0; e < 8; ++e) {
      int i = q0 + wave * 16 + mh + e;
      float v = of[dn][e] / lrow[e];
      O[((size_t)(b * SEQ + i)) * DMODEL + h * HEADD + dn * 16 + lr] = (__bf16)v;
    }
}

extern "C" void kernel_launch(void* const* d_in, const int* in_sizes, int n_in,
                              void* d_out, int out_size, void* d_ws, size_t ws_size,
                              hipStream_t stream)
{
  (void)in_sizes; (void)n_in; (void)out_size; (void)ws_size;
  const float* q    = (const float*)d_in[0];
  const float* k    = (const float*)d_in[1];
  const float* v    = (const float*)d_in[2];
  /* d_in[3] = causal mask, reproduced analytically */
  const float* wq_w = (const float*)d_in[4];
  const float* wq_b = (const float*)d_in[5];
  const float* wk_w = (const float*)d_in[6];
  const float* wk_b = (const float*)d_in[7];
  const float* wv_w = (const float*)d_in[8];
  const float* wv_b = (const float*)d_in[9];
  const float* ow_w = (const float*)d_in[10];
  const float* ow_b = (const float*)d_in[11];
  float* out = (float*)d_out;

  const size_t NQ = (size_t)BATCH * SEQ * DMODEL;   // 4M elements
  const size_t NW = (size_t)DMODEL * DMODEL;        // 1M elements
  unsigned short* wsu = (unsigned short*)d_ws;
  __bf16* qb  = (__bf16*)(wsu);
  __bf16* kb  = (__bf16*)(wsu + NQ);
  __bf16* vb  = (__bf16*)(wsu + 2 * NQ);
  __bf16* wqb = (__bf16*)(wsu + 3 * NQ);
  __bf16* wkb = (__bf16*)(wsu + 3 * NQ + NW);
  __bf16* wvb = (__bf16*)(wsu + 3 * NQ + 2 * NW);
  __bf16* owb = (__bf16*)(wsu + 3 * NQ + 3 * NW);
  __bf16* Qp  = (__bf16*)(wsu + 3 * NQ + 4 * NW);
  __bf16* Kp  = (__bf16*)(wsu + 4 * NQ + 4 * NW);
  __bf16* Vp  = (__bf16*)(wsu + 5 * NQ + 4 * NW);
  __bf16* Ob  = (__bf16*)(wsu + 6 * NQ + 4 * NW);

  const unsigned gq = (unsigned)((NQ + 255) / 256);
  const unsigned gw = (unsigned)((NW + 255) / 256);
  cvt_f32_bf16<<<gq, 256, 0, stream>>>(q,    qb,  (int)NQ);
  cvt_f32_bf16<<<gq, 256, 0, stream>>>(k,    kb,  (int)NQ);
  cvt_f32_bf16<<<gq, 256, 0, stream>>>(v,    vb,  (int)NQ);
  cvt_f32_bf16<<<gw, 256, 0, stream>>>(wq_w, wqb, (int)NW);
  cvt_f32_bf16<<<gw, 256, 0, stream>>>(wk_w, wkb, (int)NW);
  cvt_f32_bf16<<<gw, 256, 0, stream>>>(wv_w, wvb, (int)NW);
  cvt_f32_bf16<<<gw, 256, 0, stream>>>(ow_w, owb, (int)NW);

  dim3 gg(MGEMM / 128, NGEMM / 64);          // 32 x 16
  gemm_bf16<0><<<gg, 256, 0, stream>>>(qb, wqb, wq_b, Qp, nullptr);
  gemm_bf16<0><<<gg, 256, 0, stream>>>(kb, wkb, wk_b, Kp, nullptr);
  gemm_bf16<0><<<gg, 256, 0, stream>>>(vb, wvb, wv_b, Vp, nullptr);

  dim3 ga(SEQ / 128, BATCH * N_HEADS);       // 16 x 32
  attn_alibi<<<ga, 256, 0, stream>>>(Qp, Kp, Vp, Ob);

  gemm_bf16<1><<<gg, 256, 0, stream>>>(Ob, owb, ow_b, nullptr, out);
}
